// EuclideanDeconf_32899449487511
// MI455X (gfx1250) — matrix operand order repaired and checked by reading, not compiled, not run on hardware
//
#include <hip/hip_runtime.h>
#include <hip/hip_bf16.h>
#include <math.h>

typedef __attribute__((ext_vector_type(16))) __bf16 v16bf;
typedef __attribute__((ext_vector_type(8)))  __bf16 v8bf;
typedef __attribute__((ext_vector_type(8)))  float  v8f;

union AFrag {
    v16bf v;
    v8bf  h[2];
};

// ---------------------------------------------------------------------------
// CDNA5 async global->LDS copy (ASYNCcnt-tracked), GV addressing mode:
//   global_load_async_to_lds_b128  vDstLdsAddr, v[addrLo:addrHi], off
// LDS byte address = low 32 bits of the generic pointer (LDS aperture).
// ---------------------------------------------------------------------------
__device__ __forceinline__ void async_ld_b128(const void* g, void* l) {
    unsigned lds_addr = (unsigned)(uintptr_t)l;
    asm volatile("global_load_async_to_lds_b128 %0, %1, off"
                 :
                 : "v"(lds_addr), "v"(g)
                 : "memory");
}

__device__ __forceinline__ void wait_async0() {
#if __has_builtin(__builtin_amdgcn_s_wait_asynccnt)
    __builtin_amdgcn_s_wait_asynccnt(0);
#else
    asm volatile("s_wait_asynccnt 0x0" ::: "memory");
#endif
}

__device__ __forceinline__ float dot4sq(float4 a) {
    return a.x*a.x + a.y*a.y + a.z*a.z + a.w*a.w;
}

// ---------------------------------------------------------------------------
// prep_w: L2-normalize weight rows -> bf16 wn (padded to CP rows), wsq,
//         and copy raw weight into the output tail. One wave per row.
// ---------------------------------------------------------------------------
__global__ __launch_bounds__(256) void prep_w_kernel(
    const float* __restrict__ w,
    unsigned short* __restrict__ wn_us,   // [CP, 256] bf16
    float* __restrict__ wsq,              // [CP]
    float* __restrict__ wcopy,            // [C, 256] raw weight passthrough
    int C, int CP)
{
    const int lane = threadIdx.x & 31;
    const int wave = threadIdx.x >> 5;
    const int c = blockIdx.x * 8 + wave;
    if (c >= CP) return;

    __bf16* orow = (__bf16*)wn_us + (size_t)c * 256;

    if (c >= C) {
        v8bf z;
        #pragma unroll
        for (int i = 0; i < 8; ++i) z[i] = (__bf16)0.0f;
        *(v8bf*)(orow + lane * 8) = z;
        if (lane == 0) wsq[c] = 0.0f;
        return;
    }

    const float* row = w + (size_t)c * 256;
    float4 f0 = *(const float4*)(row + lane * 8);
    float4 f1 = *(const float4*)(row + lane * 8 + 4);
    float s = dot4sq(f0) + dot4sq(f1);
    #pragma unroll
    for (int o = 16; o >= 1; o >>= 1) s += __shfl_xor(s, o, 32);

    const float scale = 1.0f / fmaxf(sqrtf(s), 1e-12f);

    v8bf o8;
    o8[0] = (__bf16)(f0.x * scale); o8[1] = (__bf16)(f0.y * scale);
    o8[2] = (__bf16)(f0.z * scale); o8[3] = (__bf16)(f0.w * scale);
    o8[4] = (__bf16)(f1.x * scale); o8[5] = (__bf16)(f1.y * scale);
    o8[6] = (__bf16)(f1.z * scale); o8[7] = (__bf16)(f1.w * scale);
    *(v8bf*)(orow + lane * 8) = o8;

    *(float4*)(wcopy + (size_t)c * 256 + lane * 8)     = f0;
    *(float4*)(wcopy + (size_t)c * 256 + lane * 8 + 4) = f1;

    if (lane == 0) wsq[c] = s * scale * scale;
}

// ---------------------------------------------------------------------------
// fused_gemm: per block = 8 waves = 128 rows of x.
//   - pair 0 of wn (32 cols) prefetched to LDS via async-to-LDS at entry
//   - normalize x rows in-register (bf16 A fragments resident, K=256)
//   - loop column-pairs: double-buffered LDS (async DMA for p+1 overlaps WMMA
//     of p); TWO independent accumulator chains interleave WMMA issue
//   - fused epilogue: logits = -max(xsq + wsq - 2*dot, 0); with CC compile-time
//     the row stores are base + immediate offsets.
// CC = compile-time class count (0 -> dynamic).
// ---------------------------------------------------------------------------
template <int CC>
__global__ __launch_bounds__(256) void fused_gemm_kernel(
    const float* __restrict__ x,
    const unsigned short* __restrict__ wn_us,
    const float* __restrict__ wsq,
    float* __restrict__ logits,
    int Cdyn, int nPairs)
{
    const int C = (CC > 0) ? CC : Cdyn;

    __shared__ __bf16 Bs[2][2][16 * 264];   // [buffer][subtile]; stride 264 anti-conflict
    __shared__ float  xqs[8][16];

    const __bf16* wn = (const __bf16*)wn_us;

    const int lane = threadIdx.x & 31;
    const int wave = threadIdx.x >> 5;
    const int half = lane >> 4;          // 0 or 1
    const int l15  = lane & 15;
    const size_t rowbase = (size_t)blockIdx.x * 128 + (size_t)wave * 16;

    // Staging: one pair = 2 subtiles x 16 rows x 512B = 16 KB = 1024 b128 chunks,
    // 4 per thread.
    int goff[4], loff[4];
    #pragma unroll
    for (int i = 0; i < 4; ++i) {
        const int ch   = threadIdx.x * 4 + i;
        const int sub  = ch >> 9;            // 512 chunks per subtile
        const int rem  = ch & 511;
        const int row  = rem >> 5;           // 32 chunks per row
        const int col8 = (rem & 31) * 8;
        goff[i] = (sub * 16 + row) * 256 + col8;      // elements within a pair
        loff[i] = sub * (16 * 264) + row * 264 + col8; // padded LDS layout
    }

    // Prefetch pair 0 immediately: DMA runs during the normalization phase.
    #pragma unroll
    for (int i = 0; i < 4; ++i)
        async_ld_b128(wn + goff[i], &Bs[0][0][0] + loff[i]);

    // Row l15 of this wave's 16x256 A tile lives entirely in lanes {l15, l15+16}.
    const float* xrow = x + (rowbase + l15) * 256;

    // Pass 1: sum of squares (partial per lane, merge with lane^16).
    float ss = 0.0f;
    #pragma unroll
    for (int st = 0; st < 8; ++st) {
        const int kb = st * 32 + half * 8;
        float4 u0 = *(const float4*)(xrow + kb);
        float4 u1 = *(const float4*)(xrow + kb + 4);
        float4 u2 = *(const float4*)(xrow + kb + 16);
        float4 u3 = *(const float4*)(xrow + kb + 20);
        ss += dot4sq(u0) + dot4sq(u1) + dot4sq(u2) + dot4sq(u3);
    }
    ss += __shfl_xor(ss, 16, 32);
    const float scale = 1.0f / fmaxf(sqrtf(ss), 1e-12f);

    // Pass 2 (L2-hot reload): resident bf16 A fragments, K = 256 in 8 steps.
    // 16-bit A 16x32 layout: lane<16 holds K {kb..kb+7, kb+16..kb+23}, lane>=16 +8.
    AFrag a[8];
    #pragma unroll
    for (int st = 0; st < 8; ++st) {
        const int kb = st * 32 + half * 8;
        float4 u0 = *(const float4*)(xrow + kb);
        float4 u1 = *(const float4*)(xrow + kb + 4);
        float4 u2 = *(const float4*)(xrow + kb + 16);
        float4 u3 = *(const float4*)(xrow + kb + 20);
        v16bf av;
        av[0]  = (__bf16)(u0.x * scale); av[1]  = (__bf16)(u0.y * scale);
        av[2]  = (__bf16)(u0.z * scale); av[3]  = (__bf16)(u0.w * scale);
        av[4]  = (__bf16)(u1.x * scale); av[5]  = (__bf16)(u1.y * scale);
        av[6]  = (__bf16)(u1.z * scale); av[7]  = (__bf16)(u1.w * scale);
        av[8]  = (__bf16)(u2.x * scale); av[9]  = (__bf16)(u2.y * scale);
        av[10] = (__bf16)(u2.z * scale); av[11] = (__bf16)(u2.w * scale);
        av[12] = (__bf16)(u3.x * scale); av[13] = (__bf16)(u3.y * scale);
        av[14] = (__bf16)(u3.z * scale); av[15] = (__bf16)(u3.w * scale);
        a[st].v = av;
    }

    // Exchange xsq values: epilogue lane needs rows r + 8*half, not row l15.
    if (lane < 16) xqs[wave][l15] = ss * scale * scale;
    __syncthreads();
    float xq[8];
    #pragma unroll
    for (int r = 0; r < 8; ++r) xq[r] = xqs[wave][r + half * 8];

    // Per-lane output base: row (rowbase + half*8), col l15. With CC constant,
    // r*C folds into the store immediate; p advances a single pointer.
    float* obase = logits + (rowbase + (size_t)half * 8) * (size_t)C + l15;

    for (int p = 0; p < nPairs; ++p) {
        wait_async0();       // this wave's share of pair p resident in LDS
        __syncthreads();     // everyone's share resident; prior readers done
        if (p + 1 < nPairs) {
            const __bf16* g = wn + (size_t)(p + 1) * (32 * 256);
            __bf16* dst = &Bs[(p + 1) & 1][0][0];
            #pragma unroll
            for (int i = 0; i < 4; ++i)
                async_ld_b128(g + goff[i], dst + loff[i]);
        }
        const __bf16* B0 = &Bs[p & 1][0][0];
        const __bf16* B1 = &Bs[p & 1][1][0];

        // Two independent WMMA chains (acc0/acc1) interleave to hide the
        // WMMA->WMMA accumulator RAW latency.
        // B (KxN) striping: lane = column n, half h holds K {8h..8h+7, 16+8h..16+8h+7}.
        v8f acc0 = {}, acc1 = {};
        #pragma unroll
        for (int st = 0; st < 8; ++st) {
            const int bo = l15 * 264 + st * 32 + half * 8;
            AFrag b0, b1;
            const __bf16* bp0 = B0 + bo;
            b0.h[0] = *(const v8bf*)bp0;
            b0.h[1] = *(const v8bf*)(bp0 + 16);
            const __bf16* bp1 = B1 + bo;
            b1.h[0] = *(const v8bf*)bp1;
            b1.h[1] = *(const v8bf*)(bp1 + 16);
            acc0 = __builtin_amdgcn_wmma_f32_16x16x32_bf16(
                false, a[st].v, false, b0.v, (short)0, acc0, false, false);
            acc1 = __builtin_amdgcn_wmma_f32_16x16x32_bf16(
                false, a[st].v, false, b1.v, (short)0, acc1, false, false);
        }

        // C/D layout: VGPR r, lane l -> M = r + 8*(l>=16), N = l & 15.
        #pragma unroll
        for (int sub = 0; sub < 2; ++sub) {
            const int col = p * 32 + sub * 16 + l15;
            const v8f& acc = sub ? acc1 : acc0;
            if (col < C) {
                const float wq = wsq[col];
                float* op = obase + (size_t)p * 32 + sub * 16;
                #pragma unroll
                for (int r = 0; r < 8; ++r) {
                    float d = fmaxf(xq[r] + wq - 2.0f * acc[r], 0.0f);
                    op[(size_t)r * C] = -d;
                }
            }
        }
    }
}

// ---------------------------------------------------------------------------
extern "C" void kernel_launch(void* const* d_in, const int* in_sizes, int n_in,
                              void* d_out, int out_size, void* d_ws, size_t ws_size,
                              hipStream_t stream) {
    const float* x = (const float*)d_in[0];
    const float* w = (const float*)d_in[1];

    const int  D  = 256;
    const long Bn = (long)in_sizes[0] / D;   // 131072
    const int  C  = in_sizes[1] / D;         // 1000
    const int  nPairs = (C + 31) / 32;       // 32
    const int  CP = nPairs * 32;             // 1024 (padded classes)

    // Workspace: wn bf16 [CP,256] (512 KB) + wsq [CP]
    unsigned short* wn  = (unsigned short*)d_ws;
    float*          wsq = (float*)((char*)d_ws + (size_t)CP * D * sizeof(unsigned short));

    float* logits = (float*)d_out;
    float* wcopy  = logits + (size_t)Bn * C;   // tuple element 2: raw weight

    prep_w_kernel<<<(CP + 7) / 8, 256, 0, stream>>>(w, wn, wsq, wcopy, C, CP);
    if (C == 1000) {
        fused_gemm_kernel<1000><<<(int)(Bn / 128), 256, 0, stream>>>(
            x, wn, wsq, logits, C, nPairs);
    } else {
        fused_gemm_kernel<0><<<(int)(Bn / 128), 256, 0, stream>>>(
            x, wn, wsq, logits, C, nPairs);
    }
}
